// SequencePairEncoder_7687991460605
// MI455X (gfx1250) — compile-verified, matrix operand order checked by reading, not run on hardware
//
#include <hip/hip_runtime.h>
#include <hip/hip_bf16.h>

typedef __attribute__((ext_vector_type(16))) _Float16 v16h;
typedef __attribute__((ext_vector_type(8)))  _Float16 v8h;
typedef __attribute__((ext_vector_type(8)))  float    v8f;

union ABfrag { v16h v; v8h h[2]; };

#define BNSCALE 0.9999950000374997f  // 1/sqrt(1+1e-5)

// ---------------------------------------------------------------------------
// Weight pre-swizzle: OIHW fp32 -> fragment-ready f16 [nTile][kBlk][lane][16]
// K index k = tap*CinEff + ci (tap = ky*3+kx). Per WMMA 16-bit operand layout:
//   lane<16 : halves 0..7 -> K 0..7,  halves 8..15 -> K 16..23
//   lane>=16: halves 0..7 -> K 8..15, halves 8..15 -> K 24..31
// remap!=0 handles the fuse0 concat reorder: dest ci [f1|f2|corr|pad]
// maps to source channel [f1(0..63)|corr(64..144)|f2(145..208)].
// ---------------------------------------------------------------------------
__global__ __launch_bounds__(256) void pack_kernel(
    const float* __restrict__ w, _Float16* __restrict__ dst,
    int CinSrc, int CinEff, int Kblocks, int total, int remap)
{
  int idx = blockIdx.x * 256 + threadIdx.x;
  if (idx >= total) return;
  int perTile = Kblocks << 9;          // Kblocks*512
  int ntile = idx / perTile;
  int rem   = idx - ntile * perTile;
  int kb    = rem >> 9;
  int r2    = rem & 511;
  int lane  = r2 >> 4;
  int h     = r2 & 15;
  int k = kb * 32 + (h & 7) + ((h >> 3) << 4) + ((lane >> 4) << 3);
  int n = (ntile << 4) + (lane & 15);
  float val = 0.f;
  int Ktrue = 9 * CinEff;
  if (k < Ktrue) {
    int tap = k / CinEff;
    int ci  = k - tap * CinEff;
    int src = ci;
    bool valid;
    if (remap) {
      valid = (ci < 209);
      src = (ci < 64) ? ci : ((ci < 128) ? ci + 81 : ci - 64);
    } else {
      valid = (ci < CinSrc);
    }
    if (valid) val = w[(size_t)(n * CinSrc + src) * 9 + tap];
  }
  dst[idx] = (_Float16)val;
}

// ---------------------------------------------------------------------------
// First conv: 3->16, stride 2, 384->192. fp32 NCHW in, f16 NHWC out.
// ---------------------------------------------------------------------------
__global__ __launch_bounds__(256) void conv_first_kernel(
    const float* __restrict__ imgs, const float* __restrict__ w,
    const float* __restrict__ g, const float* __restrict__ b,
    _Float16* __restrict__ out)
{
  __shared__ float sw[432];
  __shared__ float sg[16], sb[16];
  for (int i = threadIdx.x; i < 432; i += 256) sw[i] = w[i];
  if (threadIdx.x < 16) {
    sg[threadIdx.x] = g[threadIdx.x] * BNSCALE;
    sb[threadIdx.x] = b[threadIdx.x];
  }
  __syncthreads();
  int img = blockIdx.z;
  int pix = blockIdx.x * 256 + threadIdx.x;   // < 192*192
  int oy = pix / 192, ox = pix - oy * 192;
  float in[27];
  for (int ci = 0; ci < 3; ++ci)
    for (int ky = 0; ky < 3; ++ky)
      for (int kx = 0; kx < 3; ++kx) {
        int iy = 2 * oy + ky - 1, ix = 2 * ox + kx - 1;
        float v = 0.f;
        if ((unsigned)iy < 384u && (unsigned)ix < 384u)
          v = imgs[(((size_t)img * 3 + ci) * 384 + iy) * 384 + ix];
        in[ci * 9 + ky * 3 + kx] = v;
      }
  _Float16* op = out + ((size_t)img * 36864 + pix) * 16;
  for (int co = 0; co < 16; ++co) {
    float acc = 0.f;
    for (int q = 0; q < 27; ++q) acc += in[q] * sw[co * 27 + q];
    float v = acc * sg[co] + sb[co];
    v = v > 0.f ? v : 0.1f * v;
    op[co] = (_Float16)v;
  }
}

// ---------------------------------------------------------------------------
// Generic implicit-GEMM conv block via WMMA. One wave per block computes a
// 16-pixel x (16*NT)-outchannel tile. CIN/STRIDE are compile-time so the
// k -> (tap, ci) mapping is shifts/mul-shifts, not a VALU division chain.
// One A fragment feeds NT independent WMMAs (two accumulators in flight).
// Fused BN + LeakyReLU. Optional fp32 NCHW side output (y of fuse1).
// ---------------------------------------------------------------------------
template <int CIN, int STRIDE, int NT>
__global__ __launch_bounds__(32) void conv_wmma_kernel(
    const _Float16* __restrict__ actIn, const _Float16* __restrict__ wpack,
    const float* __restrict__ gamma, const float* __restrict__ beta,
    _Float16* __restrict__ actOut, float* __restrict__ outF32,
    int Hin, int Win, int Hout, int Wout, int Cout, int Kblocks)
{
  constexpr int KTRUE = 9 * CIN;
  int lane = threadIdx.x;
  int tilesX = Wout >> 4;
  int tile = blockIdx.x;
  int oy = tile / tilesX;
  int ox0 = (tile - oy * tilesX) << 4;
  int img = blockIdx.z;
  int m  = lane & 15;
  int hi = lane >> 4;               // 0/1: which K-subgroup this lane holds
  int ox = ox0 + m;
  const _Float16* aBase = actIn + (size_t)img * Hin * Win * CIN;
  const _Float16* bBase0 =
      wpack + ((size_t)(blockIdx.y * NT) * Kblocks) * 512 + lane * 16;

  v8f c[NT];
  for (int nt = 0; nt < NT; ++nt) c[nt] = {};

#pragma unroll 2
  for (int kb = 0; kb < Kblocks; ++kb) {
    // A fragment: two 8-half contiguous chunks from NHWC activations
    ABfrag a;
    int kc = kb * 32 + (hi << 3);
#pragma unroll
    for (int ch = 0; ch < 2; ++ch) {
      int k = kc + ch * 16;
      v8h z = {};
      if (KTRUE % 32 == 0 || k < KTRUE) {
        int tap = k / CIN;            // compile-time CIN: shift / mul-shift
        int ci  = k - tap * CIN;
        int ty  = tap / 3;
        int iy  = oy * STRIDE + ty - 1;
        int ix  = ox * STRIDE + (tap - ty * 3) - 1;
        if ((unsigned)iy < (unsigned)Hin && (unsigned)ix < (unsigned)Win)
          z = *(const v8h*)(aBase + ((size_t)iy * Win + ix) * CIN + ci);
      }
      a.h[ch] = z;
    }
    if (kb + 1 < Kblocks)
      __builtin_prefetch(bBase0 + (size_t)(kb + 1) * 512, 0, 1);
#pragma unroll
    for (int nt = 0; nt < NT; ++nt) {
      ABfrag bf;
      bf.v = *(const v16h*)(bBase0 + ((size_t)nt * Kblocks + kb) * 512);
      c[nt] = __builtin_amdgcn_wmma_f32_16x16x32_f16(
          false, a.v, false, bf.v, (short)0, c[nt], false, false);
    }
  }

  // Epilogue: C/D layout lane<16 -> N=lane, M=r ; lane>=16 -> N=lane-16, M=r+8
  int n = lane & 15;
  size_t imgHW = (size_t)Hout * Wout;
#pragma unroll
  for (int nt = 0; nt < NT; ++nt) {
    int co0 = (blockIdx.y * NT + nt) << 4;
    float sc = gamma[co0 + n] * BNSCALE;
    float bt = beta[co0 + n];
#pragma unroll
    for (int r = 0; r < 8; ++r) {
      int mm = r + (hi << 3);
      float v = c[nt][r] * sc + bt;
      v = v > 0.f ? v : 0.1f * v;
      int px = oy * Wout + ox0 + mm;
      actOut[((size_t)img * imgHW + px) * Cout + co0 + n] = (_Float16)v;
      if (outF32)
        outF32[(size_t)img * Cout * imgHW + (size_t)(co0 + n) * imgHW + px] = v;
    }
  }
}

// ---------------------------------------------------------------------------
// L2 normalize feature vectors (per pixel, 64 ch).
// ---------------------------------------------------------------------------
__global__ __launch_bounds__(256) void l2norm_kernel(
    const _Float16* __restrict__ feats, _Float16* __restrict__ fnorm)
{
  int t = blockIdx.x * 256 + threadIdx.x;
  if (t >= 36 * 2304) return;
  const _Float16* p = feats + (size_t)t * 64;
  v8h chunks[8];
  float s = 0.f;
  for (int i = 0; i < 8; ++i) {
    chunks[i] = *(const v8h*)(p + i * 8);
    for (int j = 0; j < 8; ++j) { float f = (float)chunks[i][j]; s += f * f; }
  }
  float inv = 1.f / fmaxf(sqrtf(s), 1e-12f);
  _Float16* q = fnorm + (size_t)t * 64;
  for (int i = 0; i < 8; ++i) {
    v8h o;
    for (int j = 0; j < 8; ++j) o[j] = (_Float16)((float)chunks[i][j] * inv);
    *(v8h*)(q + i * 8) = o;
  }
}

// ---------------------------------------------------------------------------
// Concat f1|f2 raw features into padded 224-ch buffer (corr fills 128..208).
// ---------------------------------------------------------------------------
__global__ __launch_bounds__(256) void concat_kernel(
    const _Float16* __restrict__ feats, _Float16* __restrict__ cat)
{
  int t = blockIdx.x * 256 + threadIdx.x;
  if (t >= 32 * 2304) return;
  int pair = t / 2304, pix = t - pair * 2304;
  int b = pair >> 3, tt = pair & 7;
  const _Float16* p1 = feats + ((size_t)(b * 9 + tt) * 2304 + pix) * 64;
  const _Float16* p2 = p1 + (size_t)2304 * 64;
  _Float16* q = cat + ((size_t)pair * 2304 + pix) * 224;
  for (int i = 0; i < 8; ++i) *(v8h*)(q + i * 8)      = *(const v8h*)(p1 + i * 8);
  for (int i = 0; i < 8; ++i) *(v8h*)(q + 64 + i * 8) = *(const v8h*)(p2 + i * 8);
  for (int ci = 209; ci < 224; ++ci) q[ci] = (_Float16)0.f;
}

// ---------------------------------------------------------------------------
// 9x9 local correlation of L2-normalized features. Writes fp32 NCHW to d_out
// and f16 into the concat buffer (channels 128..208).
// ---------------------------------------------------------------------------
__global__ __launch_bounds__(256) void corr_kernel(
    const _Float16* __restrict__ fnorm, float* __restrict__ corrOut,
    _Float16* __restrict__ cat)
{
  int t = blockIdx.x * 256 + threadIdx.x;
  if (t >= 32 * 2304) return;
  int pair = t / 2304, pix = t - pair * 2304;
  int b = pair >> 3, tt = pair & 7;
  int y = pix / 48, x = pix - y * 48;
  const _Float16* p1 = fnorm + ((size_t)(b * 9 + tt) * 2304 + pix) * 64;
  v8h a[8];
  for (int i = 0; i < 8; ++i) a[i] = *(const v8h*)(p1 + i * 8);
  const _Float16* base2 = fnorm + (size_t)(b * 9 + tt + 1) * 2304 * 64;
  _Float16* qc = cat + ((size_t)pair * 2304 + pix) * 224 + 128;
  for (int s = 0; s < 81; ++s) {
    int dy = s / 9 - 4, dx = s - (s / 9) * 9 - 4;
    int iy = y + dy, ix = x + dx;
    float acc = 0.f;
    if ((unsigned)iy < 48u && (unsigned)ix < 48u) {
      const _Float16* p2 = base2 + (size_t)(iy * 48 + ix) * 64;
      for (int i = 0; i < 8; ++i) {
        v8h bv = *(const v8h*)(p2 + i * 8);
        for (int j = 0; j < 8; ++j) acc += (float)a[i][j] * (float)bv[j];
      }
    }
    corrOut[((size_t)pair * 81 + s) * 2304 + pix] = acc;
    qc[s] = (_Float16)acc;
  }
}

// ---------------------------------------------------------------------------
// Final flow conv: 64->2 with bias, fp32 NCHW output.
// ---------------------------------------------------------------------------
__global__ __launch_bounds__(256) void flow_kernel(
    const _Float16* __restrict__ z, const float* __restrict__ w,
    const float* __restrict__ bias, float* __restrict__ out)
{
  __shared__ float sw[1152];
  for (int i = threadIdx.x; i < 1152; i += 256) sw[i] = w[i];
  __syncthreads();
  int t = blockIdx.x * 256 + threadIdx.x;
  if (t >= 32 * 2304) return;
  int pair = t / 2304, pix = t - pair * 2304;
  int y = pix / 48, x = pix - y * 48;
  const _Float16* base = z + (size_t)pair * 2304 * 64;
  float a0 = bias[0], a1 = bias[1];
  for (int tap = 0; tap < 9; ++tap) {
    int iy = y + tap / 3 - 1, ix = x + tap - (tap / 3) * 3 - 1;
    if ((unsigned)iy >= 48u || (unsigned)ix >= 48u) continue;
    const _Float16* p = base + (size_t)(iy * 48 + ix) * 64;
    for (int ci = 0; ci < 64; ++ci) {
      float f = (float)p[ci];
      a0 += f * sw[ci * 9 + tap];
      a1 += f * sw[(64 + ci) * 9 + tap];
    }
  }
  out[(size_t)(pair * 2) * 2304 + pix]     = a0;
  out[(size_t)(pair * 2 + 1) * 2304 + pix] = a1;
}

// ---------------------------------------------------------------------------
extern "C" void kernel_launch(void* const* d_in, const int* in_sizes, int n_in,
                              void* d_out, int out_size, void* d_ws, size_t ws_size,
                              hipStream_t stream) {
  (void)in_sizes; (void)n_in; (void)out_size; (void)ws_size;
  const float* imgs = (const float*)d_in[0];
  const float* bbw[6]; const float* bbg[6]; const float* bbb[6];
  for (int i = 0; i < 6; ++i) {
    bbw[i] = (const float*)d_in[1 + 3 * i];
    bbg[i] = (const float*)d_in[2 + 3 * i];
    bbb[i] = (const float*)d_in[3 + 3 * i];
  }
  const float* fw0 = (const float*)d_in[19]; const float* fg0 = (const float*)d_in[20]; const float* fb0 = (const float*)d_in[21];
  const float* fw1 = (const float*)d_in[22]; const float* fg1 = (const float*)d_in[23]; const float* fb1 = (const float*)d_in[24];
  const float* hw0 = (const float*)d_in[25]; const float* hg0 = (const float*)d_in[26]; const float* hb0 = (const float*)d_in[27];
  const float* hw1 = (const float*)d_in[28]; const float* hbias = (const float*)d_in[29];

  float* outF = (float*)d_out;
  float* yOut    = outF;                 // [32,128,48,48]
  float* flowOut = outF + 9437184;       // [32,2,48,48]
  float* corrOut = outF + 9584640;       // [32,81,48,48]

  // Workspace (halves)
  _Float16* bufA  = (_Float16*)d_ws;           // 21,233,664
  _Float16* bufB  = bufA  + 21233664;          // 21,233,664
  _Float16* feats = bufB  + 21233664;          // 5,308,416
  _Float16* fnorm = feats + 5308416;           // 5,308,416
  _Float16* wpk   = fnorm + 5308416;           // 551,424

  // Packed weight offsets (halves): Kblocks*512*(Cout/16)
  _Float16* w_bb1  = wpk + 0;        // Kb=5,  Cout=16  -> 2560
  _Float16* w_bb2  = wpk + 2560;     // Kb=5,  Cout=32  -> 5120
  _Float16* w_bb3  = wpk + 7680;     // Kb=9,  Cout=32  -> 9216
  _Float16* w_bb4  = wpk + 16896;    // Kb=9,  Cout=64  -> 18432
  _Float16* w_bb5  = wpk + 35328;    // Kb=18, Cout=64  -> 36864
  _Float16* w_f0   = wpk + 72192;    // Kb=63, Cout=128 -> 258048
  _Float16* w_f1   = wpk + 330240;   // Kb=36, Cout=128 -> 147456
  _Float16* w_h0   = wpk + 477696;   // Kb=36, Cout=64  -> 73728

  // --- pack weights ---
  struct PK { const float* w; _Float16* d; int cs, ce, co, kb, rm; };
  PK pks[8] = {
    { bbw[1], w_bb1,  16,  16,  16,  5, 0 },
    { bbw[2], w_bb2,  16,  16,  32,  5, 0 },
    { bbw[3], w_bb3,  32,  32,  32,  9, 0 },
    { bbw[4], w_bb4,  32,  32,  64,  9, 0 },
    { bbw[5], w_bb5,  64,  64,  64, 18, 0 },
    { fw0,    w_f0,  209, 224, 128, 63, 1 },
    { fw1,    w_f1,  128, 128, 128, 36, 0 },
    { hw0,    w_h0,  128, 128,  64, 36, 0 },
  };
  for (int i = 0; i < 8; ++i) {
    int total = pks[i].kb * 512 * (pks[i].co / 16);
    pack_kernel<<<dim3((total + 255) / 256), dim3(256), 0, stream>>>(
        pks[i].w, pks[i].d, pks[i].cs, pks[i].ce, pks[i].kb, total, pks[i].rm);
  }

  // --- backbone ---
  conv_first_kernel<<<dim3(144, 1, 36), dim3(256), 0, stream>>>(
      imgs, bbw[0], bbg[0], bbb[0], bufA);
  conv_wmma_kernel<16, 1, 1><<<dim3(192 * 12, 1, 36), dim3(32), 0, stream>>>(
      bufA, w_bb1, bbg[1], bbb[1], bufB, nullptr, 192, 192, 192, 192, 16, 5);
  conv_wmma_kernel<16, 2, 2><<<dim3(96 * 6, 1, 36), dim3(32), 0, stream>>>(
      bufB, w_bb2, bbg[2], bbb[2], bufA, nullptr, 192, 192, 96, 96, 32, 5);
  conv_wmma_kernel<32, 1, 2><<<dim3(96 * 6, 1, 36), dim3(32), 0, stream>>>(
      bufA, w_bb3, bbg[3], bbb[3], bufB, nullptr, 96, 96, 96, 96, 32, 9);
  conv_wmma_kernel<32, 2, 2><<<dim3(48 * 3, 2, 36), dim3(32), 0, stream>>>(
      bufB, w_bb4, bbg[4], bbb[4], bufA, nullptr, 96, 96, 48, 48, 64, 9);
  conv_wmma_kernel<64, 1, 2><<<dim3(48 * 3, 2, 36), dim3(32), 0, stream>>>(
      bufA, w_bb5, bbg[5], bbb[5], feats, nullptr, 48, 48, 48, 48, 64, 18);

  // --- correlation + concat (concat buffer lives in bufA) ---
  l2norm_kernel<<<dim3(324), dim3(256), 0, stream>>>(feats, fnorm);
  concat_kernel<<<dim3(288), dim3(256), 0, stream>>>(feats, bufA);
  corr_kernel<<<dim3(288), dim3(256), 0, stream>>>(fnorm, corrOut, bufA);

  // --- fuse + head ---
  conv_wmma_kernel<224, 1, 2><<<dim3(144, 4, 32), dim3(32), 0, stream>>>(
      bufA, w_f0, fg0, fb0, bufB, nullptr, 48, 48, 48, 48, 128, 63);
  conv_wmma_kernel<128, 1, 2><<<dim3(144, 4, 32), dim3(32), 0, stream>>>(
      bufB, w_f1, fg1, fb1, bufA, yOut, 48, 48, 48, 48, 128, 36);
  conv_wmma_kernel<128, 1, 2><<<dim3(144, 2, 32), dim3(32), 0, stream>>>(
      bufA, w_h0, hg0, hb0, bufB, nullptr, 48, 48, 48, 48, 64, 36);
  flow_kernel<<<dim3(288), dim3(256), 0, stream>>>(bufB, hw1, hbias, flowOut);
}
